// DoubleAttention_17763984736452
// MI455X (gfx1250) — compile-verified, matrix operand order checked by reading, not run on hardware
//
#include <hip/hip_runtime.h>
#include <hip/hip_bf16.h>

// ---------------------------------------------------------------------------
// DoubleAttention for MI455X (gfx1250, wave32).
// All GEMMs via v_wmma_f32_16x16x32_bf16 (bf16 in, fp32 accum).
// Factorized attention: out = channel^T @ (spatial @ V)  -> avoids q x kv attn.
// Data layouts arranged so every WMMA fragment is contiguous along K:
// fragments load as two global_load_b128 per lane (ISA 7.12.2 layout).
// GEMM uses a 2x2 wave tile (4 wmma per K-step) with software pipelining;
// requires M % 32 == 0 and N % 32 == 0 (true for every launch below).
// Workspace requirement: ~41 MB.
// ---------------------------------------------------------------------------

typedef __bf16 bf16;
typedef __attribute__((ext_vector_type(16))) __bf16 v16bf;
typedef __attribute__((ext_vector_type(8)))  __bf16 v8bf;
typedef __attribute__((ext_vector_type(8)))  float  v8f;

#define QLEN  2304          // 48*48 positions
#define CCH   320           // channels
#define DAINN 512           // heads*dim_head
#define NHEAD 8
#define DHEAD 64
#define NFILT 32
#define NBATC 2
#define NROWS (NBATC * QLEN)   // 4608
#define LN_EPS 1e-5f

// ---------------------------------------------------------------------------
// Fragment loader, contiguous-K only. ISA 7.12.2 16-bit 16x32 layout (wave32):
// lane r=lane&15 selects the major index (M for A / N for B); kh=lane>>4
// selects the K-half. Per lane the fragment is exactly two contiguous
// 8 x bf16 chunks at K offsets 8*kh and 16+8*kh -> two b128 loads.
// ---------------------------------------------------------------------------
static __device__ __forceinline__ v16bf da_load_frag(
    const bf16* __restrict__ base, long long sMaj, int maj0, int k0, int lane)
{
  const int r  = lane & 15;
  const int kh = lane >> 4;
  const bf16* p = base + (long long)(maj0 + r) * sMaj + k0;
  v8bf lo = *(const v8bf*)(p + 8 * kh);
  v8bf hi = *(const v8bf*)(p + 16 + 8 * kh);
  return __builtin_shufflevector(lo, hi, 0, 1, 2, 3, 4, 5, 6, 7,
                                         8, 9, 10, 11, 12, 13, 14, 15);
}

// C/D layout (ISA 7.12.2): VGPR i, lanes 0-15 -> M=i, lanes 16-31 -> M=i+8;
// N = lane&15.  Generic store with independent row/col strides so GEMMs can
// emit transposed outputs (keeps downstream K-reads contiguous).
static __device__ __forceinline__ void da_store_tile(
    void* __restrict__ Cv, long long cBase, long long cSM, long long cSN,
    int mode, int tm, int tn, int lane, v8f acc,
    const float* __restrict__ bias, const float* __restrict__ resid, float sv)
{
  const int r  = lane & 15;
  const int kh = lane >> 4;
  const int n  = tn + r;
  if (mode == 2) {               // final: NCHW scatter + bias + residual*scale
    float* Cp = (float*)Cv;
#pragma unroll
    for (int i = 0; i < 8; ++i) {
      int m  = tm + i + 8 * kh;          // global row = b*QLEN + p
      int b2 = m / QLEN;
      int p  = m - b2 * QLEN;
      long long idx = ((long long)(b2 * CCH + n)) * QLEN + p;
      Cp[idx] = acc[i] + bias[n] + resid[idx] * sv;
    }
  } else if (mode == 1) {        // bf16 store
    bf16* Cp = (bf16*)Cv + cBase;
#pragma unroll
    for (int i = 0; i < 8; ++i)
      Cp[(long long)(tm + i + 8 * kh) * cSM + (long long)n * cSN] = (bf16)acc[i];
  } else {                       // fp32 store
    float* Cp = (float*)Cv + cBase;
#pragma unroll
    for (int i = 0; i < 8; ++i)
      Cp[(long long)(tm + i + 8 * kh) * cSM + (long long)n * cSN] = acc[i];
  }
}

#define DA_WMMA(A_, B_, C_) \
  __builtin_amdgcn_wmma_f32_16x16x32_bf16(false, (A_), false, (B_), (short)0, (C_), false, false)

// ---------------------------------------------------------------------------
// Generic batched GEMM: D[m,n] = sum_k A[m,k] * B[k,n], K contiguous on both.
// batch index z = b*heads + h; per-operand offsets are b*OffB + h*OffH.
// 128 threads = 4 waves; each wave owns a 32x32 output tile (2x2 WMMA tiles,
// 4 accumulators). K-loop is software-pipelined: next-step fragments are in
// flight while current WMMAs execute. All guards are wave-uniform -> EXEC is
// all-ones around every WMMA (ISA 7.12 requirement), no exec save/restore.
// Requires M % 32 == 0, N % 32 == 0, K % 32 == 0.
// ---------------------------------------------------------------------------
__global__ void da_gemm_wmma(
    const bf16* __restrict__ A, long long aOffB, long long aOffH, long long aSM,
    const bf16* __restrict__ B, long long bOffB, long long bOffH, long long bSN,
    void* __restrict__ Cv, long long cOffB, long long cOffH,
    long long cSM, long long cSN,
    int M, int N, int K, int heads, int mode,
    const float* __restrict__ bias, const float* __restrict__ resid,
    const float* __restrict__ scalePtr)
{
  const int z = blockIdx.z;
  const int b = z / heads;
  const int h = z - b * heads;
  const int wave = threadIdx.x >> 5;
  const int lane = threadIdx.x & 31;

  const int tm0 = blockIdx.y * 32;
  const int tn0 = blockIdx.x * 128 + wave * 32;
  if (tm0 >= M || tn0 >= N) return;          // whole-wave uniform
  const int tm1 = tm0 + 16;
  const int tn1 = tn0 + 16;

  const bf16* Ap = A + (long long)b * aOffB + (long long)h * aOffH;
  const bf16* Bp = B + (long long)b * bOffB + (long long)h * bOffH;

  v8f acc00 = {}, acc01 = {}, acc10 = {}, acc11 = {};

  // software pipeline: prologue loads K-step 0
  v16bf fa0 = da_load_frag(Ap, aSM, tm0, 0, lane);
  v16bf fa1 = da_load_frag(Ap, aSM, tm1, 0, lane);
  v16bf fb0 = da_load_frag(Bp, bSN, tn0, 0, lane);
  v16bf fb1 = da_load_frag(Bp, bSN, tn1, 0, lane);

  int k0 = 0;
  for (; k0 + 32 < K; k0 += 32) {
    if (k0 + 64 < K) {   // two steps ahead, keep near the WGP
      __builtin_prefetch(Ap + (long long)tm0 * aSM + (k0 + 64), 0, 3);
      __builtin_prefetch(Bp + (long long)tn0 * bSN + (k0 + 64), 0, 3);
    }
    v16bf na0 = da_load_frag(Ap, aSM, tm0, k0 + 32, lane);
    v16bf na1 = da_load_frag(Ap, aSM, tm1, k0 + 32, lane);
    v16bf nb0 = da_load_frag(Bp, bSN, tn0, k0 + 32, lane);
    v16bf nb1 = da_load_frag(Bp, bSN, tn1, k0 + 32, lane);
    acc00 = DA_WMMA(fa0, fb0, acc00);
    acc01 = DA_WMMA(fa0, fb1, acc01);
    acc10 = DA_WMMA(fa1, fb0, acc10);
    acc11 = DA_WMMA(fa1, fb1, acc11);
    fa0 = na0; fa1 = na1; fb0 = nb0; fb1 = nb1;
  }
  acc00 = DA_WMMA(fa0, fb0, acc00);
  acc01 = DA_WMMA(fa0, fb1, acc01);
  acc10 = DA_WMMA(fa1, fb0, acc10);
  acc11 = DA_WMMA(fa1, fb1, acc11);

  const float sv = (mode == 2) ? scalePtr[0] : 0.f;
  const long long cBase = (long long)b * cOffB + (long long)h * cOffH;
  da_store_tile(Cv, cBase, cSM, cSN, mode, tm0, tn0, lane, acc00, bias, resid, sv);
  da_store_tile(Cv, cBase, cSM, cSN, mode, tm0, tn1, lane, acc01, bias, resid, sv);
  da_store_tile(Cv, cBase, cSM, cSN, mode, tm1, tn0, lane, acc10, bias, resid, sv);
  da_store_tile(Cv, cBase, cSM, cSN, mode, tm1, tn1, lane, acc11, bias, resid, sv);
}

// ---------------------------------------------------------------------------
// Prep: NCHW -> (row, C) transpose of hidden (bf16) and LayerNorm of encoder
// states (fp32 stats, bf16 out). One wave32 per row, 8 rows per block.
// ---------------------------------------------------------------------------
__global__ void da_prep_ln(const float* __restrict__ hid,
                           const float* __restrict__ enc,
                           const float* __restrict__ gamma,
                           const float* __restrict__ beta,
                           bf16* __restrict__ hsb, bf16* __restrict__ ehb)
{
  const int wid  = threadIdx.x >> 5;
  const int lane = threadIdx.x & 31;
  const long long row = (long long)blockIdx.x * 8 + wid;
  if (row >= NROWS) return;
  const int b = (int)(row / QLEN);
  const int p = (int)(row - (long long)b * QLEN);

  const float* hp = hid + ((long long)b * CCH) * QLEN + p;  // stride QLEN per ch
  const float* ep = enc + ((long long)b * CCH) * QLEN + p;
  bf16* ho = hsb + row * CCH;
  bf16* eo = ehb + row * CCH;

  float sum = 0.f, ss = 0.f;
  for (int c = lane; c < CCH; c += 32) {
    ho[c] = (bf16)hp[(long long)c * QLEN];
    float v = ep[(long long)c * QLEN];
    sum += v; ss += v * v;
  }
  for (int o = 16; o; o >>= 1) {
    sum += __shfl_xor(sum, o, 32);
    ss  += __shfl_xor(ss,  o, 32);
  }
  const float mean = sum * (1.f / CCH);
  const float var  = ss * (1.f / CCH) - mean * mean;
  const float rstd = rsqrtf(var + LN_EPS);
  for (int c = lane; c < CCH; c += 32)
    eo[c] = (bf16)((ep[(long long)c * QLEN] - mean) * rstd * gamma[c] + beta[c]);
}

// fp32 -> bf16 elementwise (filters; already K-contiguous on the A side)
__global__ void da_cvt_bf16(const float* __restrict__ src,
                            bf16* __restrict__ dst, int n)
{
  int i = blockIdx.x * blockDim.x + threadIdx.x;
  if (i < n) dst[i] = (bf16)src[i];
}

// fp32 [K,N] -> bf16 [N,K] transpose-convert (weights: makes B K-contiguous)
__global__ void da_cvt_bf16_t(const float* __restrict__ src,
                              bf16* __restrict__ dst, int K, int N)
{
  long long i = (long long)blockIdx.x * 256 + threadIdx.x;
  if (i >= (long long)K * N) return;
  int n = (int)(i / K);
  int k = (int)(i - (long long)n * K);
  dst[i] = (bf16)src[(long long)k * N + n];   // dst[n*K + k]
}

// spatial softmax: rows of [BH*NFILT, QLEN], softmax over kv (scaled logits)
__global__ void da_softmax_rows(const float* __restrict__ src,
                                bf16* __restrict__ dst,
                                const float* __restrict__ scalePtr)
{
  const float s = scalePtr[0];
  const long long row = blockIdx.x;
  src += row * QLEN; dst += row * QLEN;
  const int tid = threadIdx.x;
  __shared__ float red[8];

  float m = -3.4e38f;
  for (int i = tid; i < QLEN; i += 256) m = fmaxf(m, src[i] * s);
  for (int o = 16; o; o >>= 1) m = fmaxf(m, __shfl_xor(m, o, 32));
  if ((tid & 31) == 0) red[tid >> 5] = m;
  __syncthreads();
  m = red[0];
#pragma unroll
  for (int w = 1; w < 8; ++w) m = fmaxf(m, red[w]);
  __syncthreads();

  float sum = 0.f;
  for (int i = tid; i < QLEN; i += 256) sum += __expf(src[i] * s - m);
  for (int o = 16; o; o >>= 1) sum += __shfl_xor(sum, o, 32);
  if ((tid & 31) == 0) red[tid >> 5] = sum;
  __syncthreads();
  sum = 0.f;
#pragma unroll
  for (int w = 0; w < 8; ++w) sum += red[w];
  const float inv = 1.f / sum;

  for (int i = tid; i < QLEN; i += 256)
    dst[i] = (bf16)(__expf(src[i] * s - m) * inv);
}

// channel softmax over the 32-filter axis, one thread per (bh, q) column.
// Output written TRANSPOSED: CHt[(bh*QLEN + q)*NFILT + f]  (contiguous per
// thread, and K-contiguous for the downstream channel^T @ SV GEMM).
__global__ void da_softmax_cols_t(const float* __restrict__ src,
                                  bf16* __restrict__ dst,
                                  const float* __restrict__ scalePtr)
{
  const float s = scalePtr[0];
  const long long g = (long long)blockIdx.x * 256 + threadIdx.x; // 0..BH*QLEN-1
  const long long bh = g / QLEN;
  const long long q  = g - bh * QLEN;
  const float* p = src + bh * NFILT * QLEN + q;
  bf16* o = dst + g * NFILT;

  float m = -3.4e38f;
#pragma unroll
  for (int f = 0; f < NFILT; ++f) m = fmaxf(m, p[(long long)f * QLEN] * s);
  float sum = 0.f;
#pragma unroll
  for (int f = 0; f < NFILT; ++f) sum += __expf(p[(long long)f * QLEN] * s - m);
  const float inv = 1.f / sum;
#pragma unroll
  for (int f = 0; f < NFILT; ++f)
    o[f] = (bf16)(__expf(p[(long long)f * QLEN] * s - m) * inv);
}

// ---------------------------------------------------------------------------
extern "C" void kernel_launch(void* const* d_in, const int* in_sizes, int n_in,
                              void* d_out, int out_size, void* d_ws, size_t ws_size,
                              hipStream_t stream) {
  (void)in_sizes; (void)n_in; (void)out_size; (void)ws_size;

  const float* hid   = (const float*)d_in[0];
  const float* enc   = (const float*)d_in[1];
  const float* Wq    = (const float*)d_in[2];
  const float* Wk    = (const float*)d_in[3];
  const float* Wv    = (const float*)d_in[4];
  const float* Wo    = (const float*)d_in[5];
  const float* bo    = (const float*)d_in[6];
  const float* gamma = (const float*)d_in[7];
  const float* beta  = (const float*)d_in[8];
  const float* exf   = (const float*)d_in[9];
  const float* dif   = (const float*)d_in[10];
  const float* scl   = (const float*)d_in[11];
  float* out = (float*)d_out;

  char* ws = (char*)d_ws;
  size_t off = 0;
  auto take = [&](size_t bytes) -> char* {
    char* p = ws + off;
    off = (off + bytes + 255) & ~(size_t)255;
    return p;
  };
  bf16*  hsb = (bf16*)take((size_t)NROWS * CCH * 2);   // hs rows, bf16
  bf16*  ehb = (bf16*)take((size_t)NROWS * CCH * 2);   // ln(ehs) rows, bf16
  bf16*  Wqb = (bf16*)take((size_t)CCH * DAINN * 2);   // [N=512, K=320] (W^T)
  bf16*  Wkb = (bf16*)take((size_t)CCH * DAINN * 2);
  bf16*  Wvb = (bf16*)take((size_t)CCH * DAINN * 2);
  bf16*  Wob = (bf16*)take((size_t)DAINN * CCH * 2);   // [N=320, K=512] (Wo^T)
  bf16*  efb = (bf16*)take((size_t)NFILT * DAINN * 2);
  bf16*  dfb = (bf16*)take((size_t)NFILT * DAINN * 2);
  bf16*  Qb  = (bf16*)take((size_t)NROWS * DAINN * 2); // [rows, 512]
  bf16*  Kb  = (bf16*)take((size_t)NROWS * DAINN * 2); // [rows, 512]
  bf16*  Vt  = (bf16*)take((size_t)NROWS * DAINN * 2); // per-b [512, 2304]
  float* SP  = (float*)take((size_t)NBATC * NHEAD * NFILT * QLEN * 4);
  float* CH  = (float*)take((size_t)NBATC * NHEAD * NFILT * QLEN * 4);
  bf16*  SPb = (bf16*)take((size_t)NBATC * NHEAD * NFILT * QLEN * 2); // [32,2304]
  bf16*  CHt = (bf16*)take((size_t)NBATC * NHEAD * NFILT * QLEN * 2); // [2304,32]
  bf16*  SVt = (bf16*)take((size_t)NBATC * NHEAD * NFILT * DHEAD * 2);// [64,32]
  bf16*  Ob  = (bf16*)take((size_t)NROWS * DAINN * 2); // [rows, 512]

  // 1) transpose + LayerNorm + bf16 convert of activations
  da_prep_ln<<<dim3(NROWS / 8), 256, 0, stream>>>(hid, enc, gamma, beta, hsb, ehb);

  // 2) weight / filter down-conversion (weights transposed -> K-contiguous B)
  da_cvt_bf16_t<<<dim3((CCH * DAINN + 255) / 256), 256, 0, stream>>>(Wq, Wqb, CCH, DAINN);
  da_cvt_bf16_t<<<dim3((CCH * DAINN + 255) / 256), 256, 0, stream>>>(Wk, Wkb, CCH, DAINN);
  da_cvt_bf16_t<<<dim3((CCH * DAINN + 255) / 256), 256, 0, stream>>>(Wv, Wvb, CCH, DAINN);
  da_cvt_bf16_t<<<dim3((DAINN * CCH + 255) / 256), 256, 0, stream>>>(Wo, Wob, DAINN, CCH);
  da_cvt_bf16<<<dim3((NFILT * DAINN + 255) / 256), 256, 0, stream>>>(exf, efb, NFILT * DAINN);
  da_cvt_bf16<<<dim3((NFILT * DAINN + 255) / 256), 256, 0, stream>>>(dif, dfb, NFILT * DAINN);

  const long long QI  = (long long)QLEN * DAINN;  // per-batch Q/K/V/O stride
  const long long FQ  = (long long)NFILT * QLEN;  // per-(b,h) filter-logit block
  const long long SV1 = (long long)NFILT * DHEAD; // per-(b,h) SV block

  // 3) Q = hs@Wq, K = ln(ehs)@Wk (row-major out), V = ln(ehs)@Wv (TRANSPOSED out)
  da_gemm_wmma<<<dim3(DAINN / 128, NROWS / 32, 1), 128, 0, stream>>>(
      hsb, 0, 0, CCH,  Wqb, 0, 0, CCH,
      Qb, 0, 0, DAINN, 1, NROWS, DAINN, CCH, 1, 1, nullptr, nullptr, nullptr);
  da_gemm_wmma<<<dim3(DAINN / 128, NROWS / 32, 1), 128, 0, stream>>>(
      ehb, 0, 0, CCH,  Wkb, 0, 0, CCH,
      Kb, 0, 0, DAINN, 1, NROWS, DAINN, CCH, 1, 1, nullptr, nullptr, nullptr);
  da_gemm_wmma<<<dim3(DAINN / 128, QLEN / 32, NBATC), 128, 0, stream>>>(  // per-b
      ehb, (long long)QLEN * CCH, 0, CCH,  Wvb, 0, 0, CCH,
      Vt, QI, 0, /*cSM=*/1, /*cSN=*/QLEN,
      QLEN, DAINN, CCH, 1, 1, nullptr, nullptr, nullptr);

  // 4) SP[f,k] = ef_h . K_h ; CH[f,q] = df_h . Q_h   (M=32,N=2304,K=64)/(b,h)
  da_gemm_wmma<<<dim3(QLEN / 128, NFILT / 32, NBATC * NHEAD), 128, 0, stream>>>(
      efb, 0, DHEAD, DAINN,
      Kb, QI, DHEAD, DAINN,
      SP, NHEAD * FQ, FQ, QLEN, 1,
      NFILT, QLEN, DHEAD, NHEAD, 0, nullptr, nullptr, nullptr);
  da_gemm_wmma<<<dim3(QLEN / 128, NFILT / 32, NBATC * NHEAD), 128, 0, stream>>>(
      dfb, 0, DHEAD, DAINN,
      Qb, QI, DHEAD, DAINN,
      CH, NHEAD * FQ, FQ, QLEN, 1,
      NFILT, QLEN, DHEAD, NHEAD, 0, nullptr, nullptr, nullptr);

  // 5) softmaxes (scale applied to logits, read from device)
  da_softmax_rows<<<dim3(NBATC * NHEAD * NFILT), 256, 0, stream>>>(SP, SPb, scl);
  da_softmax_cols_t<<<dim3((NBATC * NHEAD * QLEN) / 256), 256, 0, stream>>>(CH, CHt, scl);

  // 6) SV = spatial @ V_h^T  (M=32,N=64,K=2304), stored TRANSPOSED [64,32]
  da_gemm_wmma<<<dim3(1, 1, NBATC * NHEAD), 128, 0, stream>>>(
      SPb, NHEAD * FQ, FQ, QLEN,
      Vt, QI, (long long)DHEAD * QLEN, QLEN,
      SVt, NHEAD * SV1, SV1, /*cSM=*/1, /*cSN=*/NFILT,
      NFILT, DHEAD, QLEN, NHEAD, 1, nullptr, nullptr, nullptr);

  // 7) O_h = channel^T @ SV  (M=2304,N=64,K=32); both operands K-contiguous
  da_gemm_wmma<<<dim3(1, QLEN / 32, NBATC * NHEAD), 128, 0, stream>>>(
      CHt, NHEAD * FQ, FQ, NFILT,
      SVt, NHEAD * SV1, SV1, NFILT,
      Ob, QI, DHEAD, DAINN, 1,
      QLEN, DHEAD, NFILT, NHEAD, 1, nullptr, nullptr, nullptr);

  // 8) out = O @ Wo + bo, scattered to NCHW, + residual*scale (fused)
  da_gemm_wmma<<<dim3((CCH + 127) / 128, NROWS / 32, 1), 128, 0, stream>>>(
      Ob, 0, 0, DAINN,  Wob, 0, 0, DAINN,
      out, 0, 0, 0, 0, NROWS, CCH, DAINN, 1, 2, bo, hid, scl);
}
